// MultiHeadAttention_19215683682911
// MI455X (gfx1250) — matrix operand order repaired and checked by reading, not compiled, not run on hardware
//
#include <hip/hip_runtime.h>
#include <hip/hip_bf16.h>

// MHA on MI455X (gfx1250): bf16 WMMA everywhere, f32 accumulation.
// ~100 GFLOP vs ~50MB traffic -> compute bound; everything fits in 192MB L2.
// Tile staging uses gfx1250 async paths: global_load_async_to_lds_b128 (ASYNCcnt)
// for GEMM tiles, tensor_load_to_lds (TDM, TENSORcnt) for attention K tiles.

#define SEQ 4096
#define DM  1024
#define NH  16
#define DH  64

typedef __attribute__((ext_vector_type(16))) __bf16 v16bf;
typedef __attribute__((ext_vector_type(8)))  __bf16 v8bf;
typedef __attribute__((ext_vector_type(8)))  float  v8f;
typedef unsigned int u32x4 __attribute__((ext_vector_type(4)));
typedef unsigned int u32x8 __attribute__((ext_vector_type(8)));

union Frag16 { v16bf v; v8bf h[2]; };

__device__ __forceinline__ unsigned short f2bf(float f) {
    unsigned u = __float_as_uint(f);
    unsigned r = u + 0x7FFFu + ((u >> 16) & 1u);   // RNE
    return (unsigned short)(r >> 16);
}
__device__ __forceinline__ float bf2f(unsigned short h) {
    return __uint_as_float(((unsigned)h) << 16);
}

// generic->LDS byte offset: flat LDS addresses carry the LDS offset in [31:0]
__device__ __forceinline__ unsigned lds_off(const void* p) {
    return (unsigned)(size_t)p;
}

// async 16B global->LDS copy per lane (ASYNCcnt)
__device__ __forceinline__ void async_copy16(unsigned lds, const void* g) {
    asm volatile("global_load_async_to_lds_b128 %0, %1, off"
                 :: "v"(lds), "v"(g) : "memory");
}
__device__ __forceinline__ void wait_async0() {
    asm volatile("s_wait_asynccnt 0x0" ::: "memory");
}

// ---------------------------------------------------------------- convert
__global__ __launch_bounds__(256)
void f32_to_bf16_kernel(const float* __restrict__ in, unsigned short* __restrict__ out, int n) {
    int i = blockIdx.x * 256 + threadIdx.x;
    if (i < n) out[i] = f2bf(in[i]);
}

// ---------------------------------------------------------------- GEMM: C = A @ B^T
// A: MxK bf16 row-major, B: NxK bf16 row-major (so B-fragment reads are contiguous).
// Block = 256 thr (8 waves), tile 128x64, BK=32 (= WMMA K). Each wave: 32x32 (2x2 WMMA tiles).
#define TS_M 128
#define TS_N 64
#define TS_K 32

template<bool OUT_BF16>
__global__ __launch_bounds__(256)
void gemm_bt_wmma(const unsigned short* __restrict__ A,
                  const unsigned short* __restrict__ B,
                  void* __restrict__ Cout, int M, int N, int K)
{
    __shared__ unsigned short As[TS_M][TS_K + 8];   // row stride 80B (16B aligned, padded)
    __shared__ unsigned short Bs[TS_N][TS_K + 8];

    const int t    = threadIdx.x;
    const int lane = t & 31;
    const int wave = t >> 5;
    const int m0   = blockIdx.y * TS_M;
    const int n0   = blockIdx.x * TS_N;
    const int wm   = (wave & 3) * 32;
    const int wn   = (wave >> 2) * 32;

    v8f c[2][2] = {};

    const int arow = t >> 1, acb = (t & 1) * 16;   // A: 128x32, 16 elems/thread
    const int brow = t >> 2, bcb = (t & 3) * 8;    // B: 64x32,  8 elems/thread

    for (int k0 = 0; k0 < K; k0 += TS_K) {
        {   // stage A tile: two async 16B copies, no VGPR round-trip
            const unsigned short* gA = A + (size_t)(m0 + arow) * K + k0 + acb;
            unsigned lA = lds_off(&As[arow][acb]);
            async_copy16(lA, gA);
            asm volatile("global_load_async_to_lds_b128 %0, %1, off offset:16"
                         :: "v"(lA), "v"(gA) : "memory");   // IOFFSET adds to both sides
            if (k0 + TS_K < K)  // global_prefetch_b8 for next K tile
                __builtin_prefetch(A + (size_t)(m0 + arow) * K + k0 + TS_K + acb, 0, 0);
        }
        {   // stage B tile: one async 16B copy
            const unsigned short* gB = B + (size_t)(n0 + brow) * K + k0 + bcb;
            async_copy16(lds_off(&Bs[brow][bcb]), gB);
        }
        wait_async0();
        __syncthreads();

        // A-frag 16x32 (ISA layout: lanes 0-15 K{0-7,16-23}, lanes 16-31 K{8-15,24-31})
        const int fr = lane & 15;
        const int kb = (lane & 16) ? 8 : 0;
        const int ks = (lane & 16) ? 8 : 0;
        Frag16 a[2], b[2];
        for (int i = 0; i < 2; ++i) {
            const unsigned short* p = &As[wm + i * 16 + fr][0];
            a[i].h[0] = *(const v8bf*)(p + kb);
            a[i].h[1] = *(const v8bf*)(p + kb + 16);
        }
        // B-frag 32x16 (lanes 0-15: K{0-7,16-23}, lanes 16-31: K{8-15,24-31}; col = lane&15)
        for (int j = 0; j < 2; ++j) {
            const unsigned short* p = &Bs[wn + j * 16 + fr][0];
            b[j].h[0] = *(const v8bf*)(p + ks);
            b[j].h[1] = *(const v8bf*)(p + ks + 16);
        }
        for (int i = 0; i < 2; ++i)
            for (int j = 0; j < 2; ++j)
                c[i][j] = __builtin_amdgcn_wmma_f32_16x16x32_bf16(
                    false, a[i].v, false, b[j].v, (short)0, c[i][j], false, false);
        __syncthreads();
    }

    // C layout: VGPR r = row r (lanes 0-15) / r+8 (lanes 16-31), col = lane&15
    const int rr0 = (lane & 16) ? 8 : 0;
    const int cc  = lane & 15;
    for (int i = 0; i < 2; ++i)
        for (int j = 0; j < 2; ++j)
            for (int r = 0; r < 8; ++r) {
                int row = m0 + wm + i * 16 + r + rr0;
                int col = n0 + wn + j * 16 + cc;
                float v = c[i][j][r];
                if (OUT_BF16) ((unsigned short*)Cout)[(size_t)row * N + col] = f2bf(v);
                else          ((float*)Cout)[(size_t)row * N + col] = v;
            }
}

// ---------------------------------------------------------------- RoPE (in-place on bf16 Q,K)
__global__ __launch_bounds__(256)
void rope_kernel(unsigned short* __restrict__ Q, unsigned short* __restrict__ Kd) {
    int idx = blockIdx.x * 256 + threadIdx.x;          // SEQ*NH*32 total
    int i = idx & 31;
    int h = (idx >> 5) & (NH - 1);
    int s = idx >> 9;
    if (s >= SEQ) return;
    float inv = __powf(10000.0f, -(float)(2 * i) / (float)DH);
    float ang = (float)s * inv;
    float sn, cs;
    __sincosf(ang, &sn, &cs);
    size_t base = (size_t)s * DM + h * DH + i;
    float a0 = bf2f(Q[base]), b0 = bf2f(Q[base + 32]);
    Q[base]      = f2bf(a0 * cs - b0 * sn);
    Q[base + 32] = f2bf(b0 * cs + a0 * sn);
    float a1 = bf2f(Kd[base]), b1 = bf2f(Kd[base + 32]);
    Kd[base]      = f2bf(a1 * cs - b1 * sn);
    Kd[base + 32] = f2bf(b1 * cs + a1 * sn);
}

// ---------------------------------------------------------------- flash attention
// grid (S/64, H), block 128 (4 waves). Wave w handles query rows q0..q0+15.
__global__ __launch_bounds__(128)
void attn_wmma(const unsigned short* __restrict__ Q,
               const unsigned short* __restrict__ Kd,
               const unsigned short* __restrict__ V,
               unsigned short* __restrict__ O)
{
    __shared__ unsigned short Ks[32][72];      // [key][dh], stride 144B (128B tile row + 16B TDM pad)
    __shared__ unsigned short VTs[64][40];     // [dh][key] transposed, stride 80B
    __shared__ unsigned short Ps[4][16][40];   // per-wave P scratch (C-layout -> A-layout)

    const int t = threadIdx.x, lane = t & 31, wave = t >> 5;
    const int h    = blockIdx.y;
    const int qblk = blockIdx.x;
    const int q0   = qblk * 64 + wave * 16;
    const int col0 = h * DH;

    // Q A-fragments (16x64 split into two 16x32 K-chunks)
    Frag16 aq[2];
    {
        const int fr = lane & 15;
        const int kb = (lane & 16) ? 8 : 0;
        const unsigned short* qp = Q + (size_t)(q0 + fr) * DM + col0;
        aq[0].h[0] = *(const v8bf*)(qp + kb);
        aq[0].h[1] = *(const v8bf*)(qp + kb + 16);
        aq[1].h[0] = *(const v8bf*)(qp + 32 + kb);
        aq[1].h[1] = *(const v8bf*)(qp + 32 + kb + 16);
    }

    float m[8], l[8];
    v8f o[4] = {};
    for (int r = 0; r < 8; ++r) { m[r] = -1e30f; l[r] = 0.0f; }

    const int nkb = qblk * 2 + 2;              // key blocks of 32 covering causal range
    for (int kb = 0; kb < nkb; ++kb) {
        const int kk = kb * 32;
        // ---- stage K tile [32 keys][64 dh] via Tensor Data Mover (one wave issues) ----
        if (wave == 0) {
            unsigned long long ga =
                (unsigned long long)(size_t)(Kd + (size_t)kk * DM + col0);
            u32x4 g0;
            g0[0] = 1u;                                   // count=1 (valid user D#)
            g0[1] = lds_off(&Ks[0][0]);                   // lds_addr
            g0[2] = (unsigned)(ga & 0xFFFFFFFFu);         // global_addr[31:0]
            g0[3] = (unsigned)((ga >> 32) & 0x01FFFFFFu)  // global_addr[56:32]
                  | (2u << 30);                           // type=2 ("image")
            u32x8 g1;
            g1[0] = (1u << 16)        // data_size=1 -> 2 bytes
                  | (1u << 20)        // pad_enable
                  | (4u << 22)        // pad_interval=4 -> every 32 DWORDs (=128B row)
                  | (3u << 25);       // pad_amount=3 -> 4 DWORDs (=16B) per row
            g1[1] = (unsigned)(DH & 0xFFFF) << 16;        // tensor_dim0=64 (bits 79:48)
            g1[2] = (unsigned)(SEQ & 0xFFFF) << 16;       // tensor_dim1=4096 (bits 111:80)
            g1[3] = (unsigned)DH << 16;                   // tile_dim0=64 (bits 127:112)
            g1[4] = 32u;                                  // tile_dim1=32 keys
            g1[5] = (unsigned)DM;                         // tensor_dim0_stride=1024
            g1[6] = 0u;
            g1[7] = 0u;
            asm volatile("tensor_load_to_lds %0, %1" :: "s"(g0), "s"(g1) : "memory");
            __builtin_amdgcn_s_wait_tensorcnt(0);
        }
        {   // stage V transposed [64 dh][32 keys] (needs register transpose)
            int row = t >> 2, cb = (t & 3) * 16;
            union { uint4 u[2]; unsigned short s[16]; } vv;
            const uint4* src = (const uint4*)(V + (size_t)(kk + row) * DM + col0 + cb);
            vv.u[0] = src[0]; vv.u[1] = src[1];
            for (int e = 0; e < 16; ++e) VTs[cb + e][row] = vv.s[e];
        }
        __syncthreads();

        // scores: two 16x16 tiles; each = chained K=32 WMMAs over dh
        const int fr = lane & 15;
        const int ks = (lane & 16) ? 8 : 0;
        v8f s[2];
        for (int j = 0; j < 2; ++j) {
            const unsigned short* kp = &Ks[j * 16 + fr][0];  // lane = key col
            Frag16 bk0, bk1;
            bk0.h[0] = *(const v8bf*)(kp + ks);
            bk0.h[1] = *(const v8bf*)(kp + ks + 16);
            bk1.h[0] = *(const v8bf*)(kp + 32 + ks);
            bk1.h[1] = *(const v8bf*)(kp + 32 + ks + 16);
            v8f z = {};
            z = __builtin_amdgcn_wmma_f32_16x16x32_bf16(false, aq[0].v, false, bk0.v, (short)0, z, false, false);
            z = __builtin_amdgcn_wmma_f32_16x16x32_bf16(false, aq[1].v, false, bk1.v, (short)0, z, false, false);
            s[j] = z;
        }

        // scale + causal mask + per-row tile max
        const int rowbase = q0 + ((lane & 16) ? 8 : 0);
        float mt[8];
        for (int r = 0; r < 8; ++r) mt[r] = -1e30f;
        for (int j = 0; j < 2; ++j) {
            int key = kk + j * 16 + (lane & 15);
            for (int r = 0; r < 8; ++r) {
                float v = s[j][r] * 0.125f;               // 1/sqrt(64)
                v = (key > rowbase + r) ? -1e30f : v;
                s[j][r] = v;
                mt[r] = fmaxf(mt[r], v);
            }
        }
        for (int off = 1; off < 16; off <<= 1)            // row spans 16 lanes of half-wave
            for (int r = 0; r < 8; ++r)
                mt[r] = fmaxf(mt[r], __shfl_xor(mt[r], off, 32));

        float alpha[8], rs[8];
        for (int r = 0; r < 8; ++r) {
            float mn = fmaxf(m[r], mt[r]);
            alpha[r] = __expf(m[r] - mn);
            m[r] = mn;
            rs[r] = 0.0f;
        }
        // P = exp(s - m), write to per-wave LDS (C-layout), accumulate row sums
        for (int j = 0; j < 2; ++j) {
            int colP = j * 16 + (lane & 15);
            int rowP = (lane & 16) ? 8 : 0;
            for (int r = 0; r < 8; ++r) {
                float p = __expf(s[j][r] - m[r]);
                rs[r] += p;
                Ps[wave][rowP + r][colP] = f2bf(p);
            }
        }
        for (int off = 1; off < 16; off <<= 1)
            for (int r = 0; r < 8; ++r)
                rs[r] += __shfl_xor(rs[r], off, 32);
        for (int r = 0; r < 8; ++r) l[r] = l[r] * alpha[r] + rs[r];
        for (int c = 0; c < 4; ++c)
            for (int r = 0; r < 8; ++r)
                o[c][r] *= alpha[r];
        __syncthreads();

        // P as A-fragment 16x32 (K = 32 keys)
        Frag16 ap;
        {
            const int kbp = (lane & 16) ? 8 : 0;
            const unsigned short* pp = &Ps[wave][lane & 15][0];
            ap.h[0] = *(const v8bf*)(pp + kbp);
            ap.h[1] = *(const v8bf*)(pp + kbp + 16);
        }
        // O += P @ V : 4 output 16-col chunks
        for (int c = 0; c < 4; ++c) {
            const unsigned short* vp = &VTs[c * 16 + fr][0];  // lane = dh col
            Frag16 bv;
            bv.h[0] = *(const v8bf*)(vp + ks);
            bv.h[1] = *(const v8bf*)(vp + ks + 16);
            o[c] = __builtin_amdgcn_wmma_f32_16x16x32_bf16(
                false, ap.v, false, bv.v, (short)0, o[c], false, false);
        }
        __syncthreads();
    }

    // normalize + store bf16
    const int rr0 = (lane & 16) ? 8 : 0;
    for (int c = 0; c < 4; ++c)
        for (int r = 0; r < 8; ++r) {
            float v = o[c][r] / l[r];
            int row = q0 + rr0 + r;
            int col = col0 + c * 16 + (lane & 15);
            O[(size_t)row * DM + col] = f2bf(v);
        }
}

// ---------------------------------------------------------------- launch
extern "C" void kernel_launch(void* const* d_in, const int* in_sizes, int n_in,
                              void* d_out, int out_size, void* d_ws, size_t ws_size,
                              hipStream_t stream) {
    (void)in_sizes; (void)n_in; (void)out_size; (void)ws_size;
    const float* x  = (const float*)d_in[0];
    const float* Wq = (const float*)d_in[1];
    const float* Wk = (const float*)d_in[2];
    const float* Wv = (const float*)d_in[3];
    const float* Wo = (const float*)d_in[4];

    const size_t SD = (size_t)SEQ * DM;   // 4M elems
    const size_t DD = (size_t)DM * DM;    // 1M elems

    unsigned short* ws  = (unsigned short*)d_ws;
    unsigned short* xb  = ws;              // bf16 copies
    unsigned short* wqb = xb  + SD;
    unsigned short* wkb = wqb + DD;
    unsigned short* wvb = wkb + DD;
    unsigned short* wob = wvb + DD;
    unsigned short* Qb  = wob + DD;
    unsigned short* Kb  = Qb  + SD;
    unsigned short* Vb  = Kb  + SD;
    unsigned short* Ob  = Vb  + SD;        // total ~48 MB bf16

    f32_to_bf16_kernel<<<(int)(SD / 256), 256, 0, stream>>>(x,  xb,  (int)SD);
    f32_to_bf16_kernel<<<(int)(DD / 256), 256, 0, stream>>>(Wq, wqb, (int)DD);
    f32_to_bf16_kernel<<<(int)(DD / 256), 256, 0, stream>>>(Wk, wkb, (int)DD);
    f32_to_bf16_kernel<<<(int)(DD / 256), 256, 0, stream>>>(Wv, wvb, (int)DD);
    f32_to_bf16_kernel<<<(int)(DD / 256), 256, 0, stream>>>(Wo, wob, (int)DD);

    dim3 gg(DM / TS_N, SEQ / TS_M);  // (16, 32)
    gemm_bt_wmma<true><<<gg, 256, 0, stream>>>(xb, wqb, Qb, SEQ, DM, DM);
    gemm_bt_wmma<true><<<gg, 256, 0, stream>>>(xb, wkb, Kb, SEQ, DM, DM);
    gemm_bt_wmma<true><<<gg, 256, 0, stream>>>(xb, wvb, Vb, SEQ, DM, DM);

    rope_kernel<<<(SEQ * NH * 32) / 256, 256, 0, stream>>>(Qb, Kb);

    dim3 ga(SEQ / 64, NH);           // (64, 16)
    attn_wmma<<<ga, 128, 0, stream>>>(Qb, Kb, Vb, Ob);

    gemm_bt_wmma<false><<<gg, 256, 0, stream>>>(Ob, wob, (float*)d_out, SEQ, DM, DM);
}
